// DescriptorMatch_52570399703477
// MI455X (gfx1250) — matrix operand order repaired and checked by reading, not compiled, not run on hardware
//
#include <hip/hip_runtime.h>
#include <cstdint>

typedef __attribute__((ext_vector_type(16))) __bf16 v16bf;
typedef __attribute__((ext_vector_type(8)))  float  v8f;

constexpr int   Bc    = 2;
constexpr int   Cc    = 128;
constexpr int   Hc    = 384;
constexpr int   Wc    = 512;
constexpr int   Nc    = 256;
constexpr int   Mc    = Hc * Wc;      // 196608
constexpr int   SLABS = Mc / 16;      // 12288 column slabs of 16
constexpr int   WGM   = 192;          // GEMM workgroups per batch (12288/192 = 64 slabs each)
constexpr int   PITCH = 136;          // hi/lo slab row pitch in bf16 units (272 B, 16B-aligned)
constexpr float INV_T = 100.0f;       // 1/TEMP

__device__ __forceinline__ float sample_plane(const float* __restrict__ p, float u, float v) {
  u = fminf(fmaxf(u, 0.0f), (float)(Wc - 1));
  v = fminf(fmaxf(v, 0.0f), (float)(Hc - 1));
  float x0f = floorf(u), y0f = floorf(v);
  int x0 = (int)x0f, y0 = (int)y0f;
  int x1 = x0 + 1; if (x1 > Wc - 1) x1 = Wc - 1;
  int y1 = y0 + 1; if (y1 > Hc - 1) y1 = Hc - 1;
  float wx = u - x0f, wy = v - y0f;
  float a = p[y0 * Wc + x0], b = p[y0 * Wc + x1];
  float c = p[y1 * Wc + x0], d = p[y1 * Wc + x1];
  float top = a + (b - a) * wx;
  float bot = c + (d - c) * wx;
  return top + (bot - top) * wy;
}

// ---------------------------------------------------------------------------
// Kernel 1: desc_src = zncc(bilinear(fmap_src, kp_src)); store f32 [B][N][C]
// ---------------------------------------------------------------------------
__global__ void k_desc_src(const float* __restrict__ kp_src,
                           const float* __restrict__ fmap_src,
                           float* __restrict__ dsrc) {
  int n = blockIdx.x, b = blockIdx.y, c = threadIdx.x;
  float u = kp_src[(b * Nc + n) * 2 + 0];
  float v = kp_src[(b * Nc + n) * 2 + 1];
  const float* plane = fmap_src + ((size_t)(b * Cc + c)) * Mc;
  float val = sample_plane(plane, u, v);

  __shared__ float red[Cc];
  red[c] = val; __syncthreads();
  for (int s = Cc / 2; s > 0; s >>= 1) { if (c < s) red[c] += red[c + s]; __syncthreads(); }
  float mean = red[0] * (1.0f / Cc);
  __syncthreads();
  float d = val - mean;
  red[c] = d * d; __syncthreads();
  for (int s = Cc / 2; s > 0; s >>= 1) { if (c < s) red[c] += red[c + s]; __syncthreads(); }
  float stdv = sqrtf(fmaxf(red[0] * (1.0f / (Cc - 1)), 1e-20f));
  dsrc[((size_t)(b * Nc + n)) * Cc + c] = d / stdv;
}

// ---------------------------------------------------------------------------
// Kernel 2: fused GEMM (split-bf16 WMMA) + column std scale + online softmax.
//   match[n,m] = (sum_c dsrc[c,n] * f[c,m]) / (C * std_m)   (mean term is 0)
// 512 threads = 16 waves; wave w owns A rows [16w,16w+16), A resident in VGPRs.
// Feature slabs (16 cols x 128 ch, 8KB f32) are DMA'd global->LDS with
// global_load_async_to_lds_b128 (double buffered), converted once to separate
// bf16 hi/lo slabs laid out so B fragments are direct ds_load_b128's.
// ---------------------------------------------------------------------------
__global__ void __launch_bounds__(512) k_gemm(const float* __restrict__ dsrc,
                                              const float* __restrict__ fmap_trg,
                                              float* __restrict__ match_out,
                                              float* __restrict__ partials) {
  const int b    = blockIdx.y;
  const int tid  = threadIdx.x;
  const int wave = tid >> 5;
  const int lane = tid & 31;
  const int half = lane >> 4;   // 0: lanes 0-15, 1: lanes 16-31
  const int l16  = lane & 15;

  __shared__ __attribute__((aligned(16))) float  sRaw[2][16 * 128]; // [buf][c][m16] raw f32
  __shared__ __attribute__((aligned(16))) __bf16 sHi[16 * PITCH];   // [m16][c] bf16 hi
  __shared__ __attribute__((aligned(16))) __bf16 sLo[16 * PITCH];   // [m16][c] bf16 lo
  __shared__ float sInv[16];                                        // 1/(C*std_m)

  // ---- resident A fragments: 4 K-steps, bf16 hi/lo (contiguous K = 32*kk + 16*half + j, matching B) ----
  v16bf a_hi[4], a_lo[4];
  {
    const int nrow = wave * 16 + l16;
    const float* arow = dsrc + ((size_t)(b * Nc + nrow)) * Cc;
#pragma unroll
    for (int kk = 0; kk < 4; ++kk) {
#pragma unroll
      for (int j = 0; j < 16; ++j) {
        int k = kk * 32 + half * 16 + j;
        float x = arow[k];
        __bf16 h = (__bf16)x;
        __bf16 l = (__bf16)(x - (float)h);
        a_hi[kk][j] = h;
        a_lo[kk][j] = l;
      }
    }
  }

  // thread -> one 16B chunk of the slab: c = tid>>2, 4 columns q*4..q*4+3
  const int cch = tid >> 2, q = tid & 3;
  const float* grow = fmap_trg + ((size_t)(b * Cc + cch)) * Mc + q * 4;

  // ---- online softmax running state: 8 rows per lane ----
  float mr[8], sr[8], sur[8], svr[8];
#pragma unroll
  for (int r = 0; r < 8; ++r) { mr[r] = -3.0e38f; sr[r] = 0.f; sur[r] = 0.f; svr[r] = 0.f; }

  // prologue: DMA first slab into buffer 0 (one async b128 per wave)
  {
    unsigned int ldst = (unsigned int)(uintptr_t)&sRaw[0][0] + (unsigned int)tid * 16u;
    unsigned long long ga = (unsigned long long)(uintptr_t)(grow + blockIdx.x * 16);
    asm volatile("global_load_async_to_lds_b128 %0, %1, off" :: "v"(ldst), "v"(ga) : "memory");
  }

  int buf = 0;
  for (int s = blockIdx.x; s < SLABS; s += WGM) {
    const int m0 = s * 16;
    asm volatile("s_wait_asynccnt 0" ::: "memory");  // this wave's DMA of slab s done
    __syncthreads();                                 // all waves' DMA visible

    // issue DMA for the next slab into the other buffer (overlaps with compute)
    const int sn = s + WGM;
    if (sn < SLABS) {
      unsigned int ldst = (unsigned int)(uintptr_t)&sRaw[buf ^ 1][0] + (unsigned int)tid * 16u;
      unsigned long long ga = (unsigned long long)(uintptr_t)(grow + sn * 16);
      asm volatile("global_load_async_to_lds_b128 %0, %1, off" :: "v"(ldst), "v"(ga) : "memory");
    }

    // convert raw f32 -> separate bf16 hi/lo slabs (done once, shared by 16 waves)
    {
      const float4 f4 = *(const float4*)&sRaw[buf][cch * 16 + q * 4];
      float fv[4] = { f4.x, f4.y, f4.z, f4.w };
#pragma unroll
      for (int i = 0; i < 4; ++i) {
        float x = fv[i];
        __bf16 h = (__bf16)x;
        __bf16 l = (__bf16)(x - (float)h);
        sHi[(q * 4 + i) * PITCH + cch] = h;
        sLo[(q * 4 + i) * PITCH + cch] = l;
      }
    }
    __syncthreads();

    // per-column inv-scale = 1/(C*std), from exact f32 staging data (wave 0)
    if (wave == 0) {
      float sum = 0.f, ss = 0.f;
      const int cbase = half * 64;
#pragma unroll 4
      for (int c = 0; c < 64; ++c) {
        float x = sRaw[buf][(cbase + c) * 16 + l16];
        sum += x; ss += x * x;
      }
      sum += __shfl_xor(sum, 16, 32);
      ss  += __shfl_xor(ss, 16, 32);
      if (half == 0) {
        float var = (ss - sum * sum * (1.0f / Cc)) * (1.0f / (Cc - 1));
        sInv[l16] = 1.0f / (Cc * sqrtf(fmaxf(var, 1e-20f)));
      }
    }
    __syncthreads();

    // ---- 16x16 tile: 4 K-steps x 3 split-products of v_wmma_f32_16x16x32_bf16 ----
    v8f acc;
#pragma unroll
    for (int r = 0; r < 8; ++r) acc[r] = 0.0f;

    const char* hrow = (const char*)sHi + l16 * (PITCH * 2) + half * 32;
    const char* lrow = (const char*)sLo + l16 * (PITCH * 2) + half * 32;
#pragma unroll
    for (int kk = 0; kk < 4; ++kk) {
      // B fragment: lane col n=l16, K = 32*kk + 16*half + [0,16): 16 contiguous bf16
      union { uint4 u[2]; v16bf v; } bhi, blo;
      const uint4* hp = (const uint4*)(hrow + kk * 64);
      const uint4* lp = (const uint4*)(lrow + kk * 64);
      bhi.u[0] = hp[0]; bhi.u[1] = hp[1];
      blo.u[0] = lp[0]; blo.u[1] = lp[1];
      // small terms first, dominant last
      acc = __builtin_amdgcn_wmma_f32_16x16x32_bf16(false, a_lo[kk], false, bhi.v, (short)0, acc, false, false);
      acc = __builtin_amdgcn_wmma_f32_16x16x32_bf16(false, a_hi[kk], false, blo.v, (short)0, acc, false, false);
      acc = __builtin_amdgcn_wmma_f32_16x16x32_bf16(false, a_hi[kk], false, bhi.v, (short)0, acc, false, false);
    }

    // ---- epilogue: scale, store match_vals, online-softmax update ----
    const float invs = sInv[l16];
    const int   mcol = m0 + l16;
    const float uco  = (float)(mcol & (Wc - 1));   // u = m % W  (W = 512)
    const float vco  = (float)(mcol >> 9);         // v = m / W
#pragma unroll
    for (int r = 0; r < 8; ++r) {
      float val = acc[r] * invs;
      int row = wave * 16 + half * 8 + r;
      match_out[(size_t)(b * Nc + row) * Mc + mcol] = val;

      float nm = fmaxf(mr[r], val);
      float sc = __expf((mr[r] - nm) * INV_T);
      float e  = __expf((val   - nm) * INV_T);
      sr[r]  = sr[r]  * sc + e;
      sur[r] = sur[r] * sc + e * uco;
      svr[r] = svr[r] * sc + e * vco;
      mr[r]  = nm;
    }
    buf ^= 1;
  }

  // ---- merge online state across the 16 columns (lanes) per half-wave ----
#pragma unroll
  for (int step = 1; step < 16; step <<= 1) {
#pragma unroll
    for (int r = 0; r < 8; ++r) {
      float om  = __shfl_xor(mr[r],  step, 32);
      float os  = __shfl_xor(sr[r],  step, 32);
      float osu = __shfl_xor(sur[r], step, 32);
      float osv = __shfl_xor(svr[r], step, 32);
      float nm = fmaxf(mr[r], om);
      float s1 = __expf((mr[r] - nm) * INV_T);
      float s2 = __expf((om    - nm) * INV_T);
      sr[r]  = sr[r]  * s1 + os  * s2;
      sur[r] = sur[r] * s1 + osu * s2;
      svr[r] = svr[r] * s1 + osv * s2;
      mr[r]  = nm;
    }
  }
  if (l16 == 0) {
#pragma unroll
    for (int r = 0; r < 8; ++r) {
      int n = wave * 16 + half * 8 + r;
      float* pp = partials + (((size_t)(b * WGM + blockIdx.x)) * Nc + n) * 4;
      pp[0] = mr[r]; pp[1] = sr[r]; pp[2] = sur[r]; pp[3] = svr[r];
    }
  }
}

// ---------------------------------------------------------------------------
// Kernel 3: deterministic merge of per-workgroup softmax partials -> kp_pseudo
// ---------------------------------------------------------------------------
__global__ void k_reduce(const float* __restrict__ partials,
                         float* __restrict__ out_kp) {
  int idx = blockIdx.x * blockDim.x + threadIdx.x;
  if (idx >= Bc * Nc) return;
  int b = idx / Nc, n = idx % Nc;
  float m = -3.0e38f, s = 0.f, su = 0.f, sv = 0.f;
  for (int g = 0; g < WGM; ++g) {
    const float* pp = partials + (((size_t)(b * WGM + g)) * Nc + n) * 4;
    float om = pp[0], os = pp[1], osu = pp[2], osv = pp[3];
    float nm = fmaxf(m, om);
    float s1 = __expf((m  - nm) * INV_T);
    float s2 = __expf((om - nm) * INV_T);
    s  = s  * s1 + os  * s2;
    su = su * s1 + osu * s2;
    sv = sv * s1 + osv * s2;
    m = nm;
  }
  out_kp[idx * 2 + 0] = su / s;
  out_kp[idx * 2 + 1] = sv / s;
}

// ---------------------------------------------------------------------------
// Kernel 4: desc_pseudo (bilinear+zncc), mvp, scores, weights
// ---------------------------------------------------------------------------
__global__ void k_final(const float* __restrict__ kp_src,
                        const float* __restrict__ fmap_trg,
                        const float* __restrict__ smap_src,
                        const float* __restrict__ smap_trg,
                        const float* __restrict__ dsrc,
                        const float* __restrict__ kp_pseudo,
                        float* __restrict__ out_w,
                        float* __restrict__ out_ss,
                        float* __restrict__ out_sp,
                        float* __restrict__ out_mvp) {
  int n = blockIdx.x, b = blockIdx.y, c = threadIdx.x;
  float u = kp_pseudo[(b * Nc + n) * 2 + 0];
  float v = kp_pseudo[(b * Nc + n) * 2 + 1];
  const float* plane = fmap_trg + ((size_t)(b * Cc + c)) * Mc;
  float val = sample_plane(plane, u, v);

  __shared__ float red[Cc];
  red[c] = val; __syncthreads();
  for (int s = Cc / 2; s > 0; s >>= 1) { if (c < s) red[c] += red[c + s]; __syncthreads(); }
  float mean = red[0] * (1.0f / Cc);
  __syncthreads();
  float d = val - mean;
  red[c] = d * d; __syncthreads();
  for (int s = Cc / 2; s > 0; s >>= 1) { if (c < s) red[c] += red[c + s]; __syncthreads(); }
  float stdv = sqrtf(fmaxf(red[0] * (1.0f / (Cc - 1)), 1e-20f));
  float dpn = d / stdv;
  __syncthreads();

  red[c] = dsrc[((size_t)(b * Nc + n)) * Cc + c] * dpn;
  __syncthreads();
  for (int s = Cc / 2; s > 0; s >>= 1) { if (c < s) red[c] += red[c + s]; __syncthreads(); }

  if (c == 0) {
    float mvp = red[0] * (1.0f / Cc);
    float us = kp_src[(b * Nc + n) * 2 + 0];
    float vs = kp_src[(b * Nc + n) * 2 + 1];
    float ss = sample_plane(smap_src + (size_t)b * Mc, us, vs);
    float sp = sample_plane(smap_trg + (size_t)b * Mc, u, v);
    int idx = b * Nc + n;
    out_ss[idx]  = ss;
    out_sp[idx]  = sp;
    out_mvp[idx] = mvp;
    out_w[idx]   = 0.5f * (mvp + 1.0f) * ss * sp;
  }
}

// ---------------------------------------------------------------------------
extern "C" void kernel_launch(void* const* d_in, const int* in_sizes, int n_in,
                              void* d_out, int out_size, void* d_ws, size_t ws_size,
                              hipStream_t stream) {
  (void)in_sizes; (void)n_in; (void)out_size; (void)ws_size;
  const float* kp_src   = (const float*)d_in[0];
  const float* fmap_src = (const float*)d_in[3];
  const float* fmap_trg = (const float*)d_in[4];
  const float* smap_src = (const float*)d_in[5];
  const float* smap_trg = (const float*)d_in[6];

  float* out       = (float*)d_out;
  float* out_kp    = out;                               // [B,N,2]  1024
  float* out_w     = out + 1024;                        // [B,1,N]   512
  float* out_ss    = out + 1536;                        // [B,1,N]   512
  float* out_sp    = out + 2048;                        // [B,1,N]   512
  float* out_match = out + 2560;                        // [B,N,M]
  float* out_mvp   = out + 2560 + (size_t)Bc * Nc * Mc; // [B,N]     512

  float* ws       = (float*)d_ws;
  float* dsrc     = ws;                         // B*N*C     = 65536 floats
  float* partials = ws + (size_t)Bc * Nc * Cc;  // B*WGM*N*4 = 393216 floats

  k_desc_src<<<dim3(Nc, Bc), Cc, 0, stream>>>(kp_src, fmap_src, dsrc);
  k_gemm<<<dim3(WGM, Bc), 512, 0, stream>>>(dsrc, fmap_trg, out_match, partials);
  k_reduce<<<(Bc * Nc + 255) / 256, 256, 0, stream>>>(partials, out_kp);
  k_final<<<dim3(Nc, Bc), Cc, 0, stream>>>(kp_src, fmap_trg, smap_src, smap_trg,
                                           dsrc, out_kp, out_w, out_ss, out_sp, out_mvp);
}
